// AdditiveAttention2D_25142738551072
// MI455X (gfx1250) — compile-verified, matrix operand order checked by reading, not run on hardware
//
#include <hip/hip_runtime.h>

typedef __attribute__((ext_vector_type(16))) _Float16 v16h;
typedef __attribute__((ext_vector_type(8)))  float    v8f;
typedef __attribute__((ext_vector_type(2)))  float    v2f;

#define BB 2
#define LL 512
#define DD 128
#define NCH 4           // key chunks (flash-style split-K of the softmax)
#define CHK (LL / NCH)  // 128 keys per chunk

// ---------------------------------------------------------------------------
// Kernel 1: Y[1024x128] = X[1024x128] @ W[128x128] in pure fp32 via
// V_WMMA_F32_16X16X4_F32.  grid.x = 64 row tiles, 256 threads = 8 waves,
// wave w owns output column tile w*16.
// ---------------------------------------------------------------------------
__global__ __launch_bounds__(256) void gemm_xw(const float* __restrict__ X,
                                               const float* __restrict__ W,
                                               float* __restrict__ Y) {
  const int lane = threadIdx.x & 31;
  const int wave = threadIdx.x >> 5;          // 0..7 -> column tile
  const int m    = lane & 15;
  const int hh   = lane >> 4;                 // 0 or 1
  const int row0 = blockIdx.x * 16;
  const int col0 = wave * 16;
  const int koff = hh * 2;

  const float* xrow = X + (size_t)(row0 + m) * DD;

  v8f acc = {};
  #pragma unroll 4
  for (int kb = 0; kb < DD; kb += 4) {
    v2f A, Bv;
    A[0]  = xrow[kb + koff];
    A[1]  = xrow[kb + koff + 1];
    Bv[0] = W[(kb + koff) * DD + col0 + m];
    Bv[1] = W[(kb + koff + 1) * DD + col0 + m];
    acc = __builtin_amdgcn_wmma_f32_16x16x4_f32(false, A, false, Bv,
                                                (short)0, acc, false, false);
  }
  #pragma unroll
  for (int g = 0; g < 8; ++g)
    Y[(size_t)(row0 + g + hh * 8) * DD + col0 + m] = acc[g];
}

// ---------------------------------------------------------------------------
// Kernel 2: ht16[b][d][j] = (f16) h[b][j][d]  (transposed so WMMA B-fragment
// loads of P@H are contiguous pairs along the key axis).
// ---------------------------------------------------------------------------
__global__ void conv_ht(const float* __restrict__ h, _Float16* __restrict__ ht) {
  int idx = blockIdx.x * 256 + threadIdx.x;
  if (idx >= BB * LL * DD) return;
  int b   = idx / (LL * DD);
  int rem = idx - b * LL * DD;
  int j   = rem >> 7;
  int d   = rem & (DD - 1);
  ht[((size_t)b * DD + d) * LL + j] = (_Float16)h[idx];
}

// ---------------------------------------------------------------------------
// Kernel 3: per-(batch, 16-query tile, 128-key chunk) partial attention:
//   scores -> chunk-local max/expsum -> unnormalized exp -> P_chunk @ H_chunk
// grid = (L/16, NCH, B), 512 threads (16 waves).
// LDS (35328 B):
//   [0,8192)       scores f32 [16][128]
//   [8192,16384)   sW_s   f32 [16][128]   } aliased later by
//   [16384,32768)  hU_s   f32 [32][128]   } p16 f16 [16][128] (first 4KB)
//   [32768,33280)  v_s    f32 [128]
//   [33280,35328)  red    f32 [16][32]
// ---------------------------------------------------------------------------
__global__ __launch_bounds__(512) void attn_part(const float* __restrict__ sW,
                                                 const float* __restrict__ hU,
                                                 const float* __restrict__ vvec,
                                                 const _Float16* __restrict__ ht,
                                                 float* __restrict__ mbuf,
                                                 float* __restrict__ sbuf,
                                                 float* __restrict__ Obuf) {
  __shared__ __align__(16) unsigned char smem[35328];
  float*    scores = (float*)(smem);
  float*    sW_s   = (float*)(smem + 8192);
  float*    hU_s   = (float*)(smem + 16384);
  float*    v_s    = (float*)(smem + 32768);
  float*    red    = (float*)(smem + 33280);
  _Float16* p16    = (_Float16*)(smem + 8192);   // reuses dead sW_s space

  const int tid   = threadIdx.x;
  const int itile = blockIdx.x;
  const int chunk = blockIdx.y;
  const int b     = blockIdx.z;
  const int i0    = itile * 16;
  const int jbase = chunk * CHK;

  for (int t = tid; t < 16 * DD; t += 512)
    sW_s[t] = sW[((size_t)b * LL + i0 + (t >> 7)) * DD + (t & 127)];
  for (int t = tid; t < DD; t += 512) v_s[t] = vvec[t];

  const int r = tid >> 5;   // query row 0..15
  const int c = tid & 31;   // key lane  0..31

  // ---- scores[i][j] = sum_d v[d] * tanh(sW[i][d] + hU[j][d]) ----
  for (int j0 = 0; j0 < CHK; j0 += 32) {
    __syncthreads();
    for (int t = tid; t < 32 * DD; t += 512)
      hU_s[t] = hU[((size_t)b * LL + jbase + j0 + (t >> 7)) * DD + (t & 127)];
    __syncthreads();
    float acc = 0.f;
    const float4* sr4 = (const float4*)(sW_s + r * DD);
    const float4* hr4 = (const float4*)(hU_s + c * DD);
    const float4* v4  = (const float4*)v_s;
    #pragma unroll 8
    for (int q = 0; q < DD / 4; ++q) {
      float4 a = sr4[q], k = hr4[q], vv = v4[q];
      acc = __builtin_fmaf(vv.x, tanhf(a.x + k.x), acc);
      acc = __builtin_fmaf(vv.y, tanhf(a.y + k.y), acc);
      acc = __builtin_fmaf(vv.z, tanhf(a.z + k.z), acc);
      acc = __builtin_fmaf(vv.w, tanhf(a.w + k.w), acc);
    }
    scores[r * CHK + j0 + c] = acc;
  }
  __syncthreads();

  // ---- chunk-local max and expsum along 128 keys (32 threads per row) ----
  float mx = -3.4e38f;
  #pragma unroll
  for (int t = 0; t < CHK / 32; ++t) mx = fmaxf(mx, scores[r * CHK + c + 32 * t]);
  red[r * 32 + c] = mx;
  __syncthreads();
  float rmax = -3.4e38f;
  #pragma unroll
  for (int k = 0; k < 32; ++k) rmax = fmaxf(rmax, red[r * 32 + k]);
  __syncthreads();
  float sum = 0.f;
  #pragma unroll
  for (int t = 0; t < CHK / 32; ++t) {
    float e = __expf(scores[r * CHK + c + 32 * t] - rmax);
    scores[r * CHK + c + 32 * t] = e;
    sum += e;
  }
  red[r * 32 + c] = sum;
  __syncthreads();
  float rsum = 0.f;
  #pragma unroll
  for (int k = 0; k < 32; ++k) rsum += red[r * 32 + k];
  __syncthreads();          // sW_s fully dead -> safe to alias with p16
  #pragma unroll
  for (int t = 0; t < CHK / 32; ++t)
    p16[r * CHK + c + 32 * t] = (_Float16)scores[r * CHK + c + 32 * t];
  if (c == 0) {
    int sb = (((b * (LL / 16) + itile) * NCH) + chunk) * 16 + r;
    mbuf[sb] = rmax;
    sbuf[sb] = rsum;
  }
  __syncthreads();

  // ---- O_chunk[16x128] = P[16x128] @ H_chunk[128x128], f16 WMMA ----
  const int wave = tid >> 5;
  const int lane = tid & 31;
  if (wave < 8) {                       // wave-uniform: EXEC all-ones inside
    const int mrow = lane & 15;
    const int hh   = lane >> 4;
    const int n    = wave * 16 + mrow;  // output feature (d) index
    const _Float16* hrow = ht + ((size_t)b * DD + n) * LL + jbase;
    const _Float16* prow = p16 + mrow * CHK;
    v8f accv = {};
    #pragma unroll
    for (int kb = 0; kb < CHK; kb += 32) {
      v16h Af, Bf;
      #pragma unroll
      for (int g = 0; g < 8; ++g) {
        int k0 = 2 * g + ((g >= 4) ? 8 : 0) + hh * 8;
        Af[2 * g]     = prow[kb + k0];
        Af[2 * g + 1] = prow[kb + k0 + 1];
        Bf[2 * g]     = hrow[kb + k0];
        Bf[2 * g + 1] = hrow[kb + k0 + 1];
      }
      accv = __builtin_amdgcn_wmma_f32_16x16x32_f16(false, Af, false, Bf,
                                                    (short)0, accv, false, false);
    }
    #pragma unroll
    for (int g = 0; g < 8; ++g)
      Obuf[(size_t)((((b * (LL / 16) + itile) * NCH) + chunk) * 16 + g + hh * 8) * DD + n]
          = accv[g];
  }
}

// ---------------------------------------------------------------------------
// Kernel 4: combine the NCH partial results per output row:
//   M = max m_k;  out = sum_k e^{m_k-M} O_k / sum_k e^{m_k-M} s_k
// ---------------------------------------------------------------------------
__global__ void combine(const float* __restrict__ mbuf, const float* __restrict__ sbuf,
                        const float* __restrict__ Obuf, float* __restrict__ out) {
  int idx = blockIdx.x * 256 + threadIdx.x;
  if (idx >= BB * LL * DD) return;
  int b   = idx / (LL * DD);
  int rem = idx - b * (LL * DD);
  int i   = rem >> 7;
  int d   = rem & 127;
  int itile = i >> 4, r = i & 15;
  int sb0 = ((b * (LL / 16) + itile) * NCH) * 16 + r;     // +16 per chunk
  float m0 = mbuf[sb0],      m1 = mbuf[sb0 + 16],
        m2 = mbuf[sb0 + 32], m3 = mbuf[sb0 + 48];
  float M  = fmaxf(fmaxf(m0, m1), fmaxf(m2, m3));
  float w0 = __expf(m0 - M), w1 = __expf(m1 - M),
        w2 = __expf(m2 - M), w3 = __expf(m3 - M);
  float S  = sbuf[sb0] * w0 + sbuf[sb0 + 16] * w1 +
             sbuf[sb0 + 32] * w2 + sbuf[sb0 + 48] * w3;
  size_t ob = (size_t)(((b * (LL / 16) + itile) * NCH) * 16 + r) * DD + d;  // +16*DD/chunk
  float o = Obuf[ob] * w0 + Obuf[ob + 16 * DD] * w1 +
            Obuf[ob + 32 * DD] * w2 + Obuf[ob + 48 * DD] * w3;
  out[idx] = o / S;
}

// ---------------------------------------------------------------------------
extern "C" void kernel_launch(void* const* d_in, const int* in_sizes, int n_in,
                              void* d_out, int out_size, void* d_ws, size_t ws_size,
                              hipStream_t stream) {
  const float* s = (const float*)d_in[0];   // [B,L,D]
  const float* h = (const float*)d_in[1];   // [B,L,D]
  const float* W = (const float*)d_in[2];   // [D,D]
  const float* U = (const float*)d_in[3];   // [D,D]
  const float* v = (const float*)d_in[4];   // [D,1]
  float* out = (float*)d_out;

  char* ws = (char*)d_ws;
  float*    sWbuf = (float*)(ws);                 // 512 KB
  float*    hUbuf = (float*)(ws + 524288);        // 512 KB
  _Float16* htbuf = (_Float16*)(ws + 1048576);    // 256 KB
  float*    mbuf  = (float*)(ws + 1310720);       // 16 KB  [B][32][NCH][16]
  float*    sbuf  = (float*)(ws + 1327104);       // 16 KB
  float*    Obuf  = (float*)(ws + 1343488);       // 2 MB   [B][32][NCH][16][128]

  gemm_xw<<<dim3((BB * LL) / 16), 256, 0, stream>>>(s, W, sWbuf);
  gemm_xw<<<dim3((BB * LL) / 16), 256, 0, stream>>>(h, U, hUbuf);
  conv_ht<<<dim3((BB * LL * DD + 255) / 256), 256, 0, stream>>>(h, htbuf);
  attn_part<<<dim3(LL / 16, NCH, BB), 512, 0, stream>>>(sWbuf, hUbuf, v, htbuf,
                                                        mbuf, sbuf, Obuf);
  combine<<<dim3((BB * LL * DD + 255) / 256), 256, 0, stream>>>(mbuf, sbuf, Obuf, out);
}
